// GridNetwork_56753697849983
// MI455X (gfx1250) — compile-verified, matrix operand order checked by reading, not run on hardware
//
#include <hip/hip_runtime.h>
#include <hip/hip_bf16.h>

// ---------------------------------------------------------------------------
// GridNetwork RNN on MI455X (gfx1250): bf16 WMMA (v_wmma_f32_16x16x32_bf16)
// Step-t launch fuses the decoder for step t-1 (runs concurrently on spare
// WGPs); weight pre-format packs bf16 pairs for 4B stores.
// ---------------------------------------------------------------------------

typedef __attribute__((ext_vector_type(16))) __bf16 v16bf;
typedef __attribute__((ext_vector_type(8)))  __bf16 v8bf;
typedef __attribute__((ext_vector_type(2)))  __bf16 v2bf;
typedef __attribute__((ext_vector_type(8)))  float  v8f;

#define SEQ   100
#define BATCH 200
#define NP    512
#define NG    4096
#define MPAD  208   // 13 * 16 (batch padded to multiple of 16)

#define STEP_BLOCKS 104   // 13 m-tiles * 64 n-tiles / 8 waves
#define DEC_BLOCKS  13    // 13 m-tiles * 8 n-tiles / 8 waves

// ---- WMMA wrapper --------------------------------------------------------
__device__ __forceinline__ v8f wmma_bf16(v16bf a, v16bf b, v8f c) {
  // (neg_a, A, neg_b, B, c_mod, C, reuse_a, reuse_b)
  return __builtin_amdgcn_wmma_f32_16x16x32_bf16(false, a, false, b,
                                                 (short)0, c, false, false);
}

// ---- A-operand loader: row-major bf16, per-ISA 16x32 A layout ------------
// Lane L (lane16=L&15, half=L>>4) holds row m=mBase+lane16;
// elements 0..7  <- k = ktile + half*8 + [0..8)
// elements 8..15 <- k = ktile + 16 + half*8 + [0..8)
__device__ __forceinline__ v16bf load_a_pair(const __bf16* __restrict__ rowBase,
                                             int k0) {
  v8bf lo = *reinterpret_cast<const v8bf*>(rowBase + k0);
  v8bf hi = *reinterpret_cast<const v8bf*>(rowBase + k0 + 16);
  v16bf r;
#pragma unroll
  for (int i = 0; i < 8; ++i) { r[i] = lo[i]; r[i + 8] = hi[i]; }
  return r;
}

// ---- Generic 16x64 output tile GEMM core ---------------------------------
// A: MPAD x KDIM row-major bf16 (hidden/activations)
// Bf: pre-formatted weights, (NDIM/16) x (KDIM/32) blocks of 512 bf16,
//     block layout: element index = lane*16 + e  (directly loadable as v16bf)
template <int KDIM>
__device__ __forceinline__ void gemm_tile(const __bf16* __restrict__ A,
                                          const __bf16* __restrict__ Bf,
                                          int mBase, int nBase, int lane,
                                          v8f acc[4]) {
  const int lane16 = lane & 15;
  const int half   = lane >> 4;
  const __bf16* aRow = A + (size_t)(mBase + lane16) * KDIM;
  const int KB = KDIM >> 5;                      // k-blocks
  const __bf16* bBase = Bf + ((size_t)(nBase >> 4) * KB) * 512 + lane * 16;
#pragma unroll 2
  for (int k = 0; k < KDIM; k += 32) {
    v16bf a = load_a_pair(aRow, k + half * 8);
    const __bf16* bBlk = bBase + (size_t)(k >> 5) * 512;
#pragma unroll
    for (int s = 0; s < 4; ++s) {
      v16bf b = *reinterpret_cast<const v16bf*>(bBlk + (size_t)s * KB * 512);
      acc[s] = wmma_bf16(a, b, acc[s]);
    }
  }
}

// ---- Weight pre-format: f32 (N x K row-major) -> bf16 WMMA-B blocks ------
// Each thread converts an (even k, odd k) pair: elements e, e+1 are adjacent
// in the block, so one 4-byte store.  For k-in-block kk (even):
//   half = (kk>>3)&1,  e = ((kk&16)>>1)|(kk&7),  lane = half*16 + (n&15)
__global__ void fmt_weight_bf16(const float* __restrict__ W,
                                __bf16* __restrict__ Wf, int N, int K) {
  int idx = blockIdx.x * blockDim.x + threadIdx.x;   // over N*K/2
  if (idx >= (N * K) >> 1) return;
  int Kh = K >> 1;
  int n  = idx / Kh;
  int k  = (idx - n * Kh) * 2;
  int kk   = k & 31;
  int half = (kk >> 3) & 1;
  int e    = ((kk & 16) >> 1) | (kk & 7);
  int lane = half * 16 + (n & 15);
  size_t blockId = (size_t)(n >> 4) * (K >> 5) + (k >> 5);
  v2bf pair;
  pair[0] = (__bf16)W[(size_t)n * K + k];
  pair[1] = (__bf16)W[(size_t)n * K + k + 1];
  *reinterpret_cast<v2bf*>(&Wf[blockId * 512 + lane * 16 + e]) = pair;
}

// ---- init_pc -> padded bf16 (zero pad rows 200..207) ---------------------
__global__ void pad_pc_bf16(const float* __restrict__ pc,
                            __bf16* __restrict__ pcb) {
  int idx = blockIdx.x * blockDim.x + threadIdx.x;
  if (idx >= MPAD * NP) return;
  int b = idx / NP, p = idx - b * NP;
  float v = (b < BATCH) ? pc[b * NP + p] : 0.0f;
  pcb[idx] = (__bf16)v;
}

// ---- Encoder: h0 = init_pc @ W_enc^T  (MPAD x NP -> MPAD x NG) -----------
__global__ __launch_bounds__(256) void encoder_kernel(
    const __bf16* __restrict__ pcb, const __bf16* __restrict__ Wencf,
    __bf16* __restrict__ h0) {
  int wave = (blockIdx.x * blockDim.x + threadIdx.x) >> 5;
  int lane = threadIdx.x & 31;
  int mtile = wave >> 6;           // NG/64 = 64 n-tiles per m-tile
  int nt    = wave & 63;
  int mBase = mtile * 16, nBase = nt * 64;
  v8f acc[4] = {};
  gemm_tile<NP>(pcb, Wencf, mBase, nBase, lane, acc);
  int lane16 = lane & 15, half = lane >> 4;
#pragma unroll
  for (int s = 0; s < 4; ++s) {
    int n = nBase + s * 16 + lane16;
#pragma unroll
    for (int r = 0; r < 8; ++r) {
      int m = mBase + half * 8 + r;
      h0[(size_t)m * NG + n] = (__bf16)acc[s][r];
    }
  }
}

// ---- Fused RNN step + previous-step decoder ------------------------------
// Blocks [0, STEP_BLOCKS):      h' = relu(vx_t + h @ W_hh^T)
//                               writes f32 g[t] and bf16 h' (next buffer)
// Blocks [STEP_BLOCKS, +DEC):   logits[t-1] = h @ W_dec^T   (skipped if t==0)
// Both halves only READ `hcur`, so they can run concurrently.
__global__ __launch_bounds__(256) void rnn_step_fused_kernel(
    const __bf16* __restrict__ hcur, const __bf16* __restrict__ Whhf,
    const __bf16* __restrict__ Wdecf, const float* __restrict__ velocity,
    const float* __restrict__ W_ih, __bf16* __restrict__ hnext,
    float* __restrict__ gout, float* __restrict__ logits, int t) {
  int lane   = threadIdx.x & 31;
  int lane16 = lane & 15, half = lane >> 4;

  if (blockIdx.x < STEP_BLOCKS) {
    // ---------------- recurrent GEMM ----------------
    int wave  = (blockIdx.x * blockDim.x + threadIdx.x) >> 5;
    int mtile = wave >> 6;
    int nt    = wave & 63;
    int mBase = mtile * 16, nBase = nt * 64;
    v8f acc[4] = {};
    gemm_tile<NG>(hcur, Whhf, mBase, nBase, lane, acc);

    // velocity for the 8 rows this lane owns
    float v0[8], v1[8];
#pragma unroll
    for (int r = 0; r < 8; ++r) {
      int m = mBase + half * 8 + r;
      if (m < BATCH) {
        const float* vp = velocity + ((size_t)t * BATCH + m) * 2;
        v0[r] = vp[0]; v1[r] = vp[1];
      } else {
        v0[r] = 0.0f; v1[r] = 0.0f;
      }
    }
#pragma unroll
    for (int s = 0; s < 4; ++s) {
      int n = nBase + s * 16 + lane16;
      float wi0 = W_ih[n * 2 + 0];
      float wi1 = W_ih[n * 2 + 1];
#pragma unroll
      for (int r = 0; r < 8; ++r) {
        int m = mBase + half * 8 + r;
        float h = acc[s][r] + v0[r] * wi0 + v1[r] * wi1;
        h = h > 0.0f ? h : 0.0f;
        hnext[(size_t)m * NG + n] = (__bf16)h;
        if (m < BATCH) gout[((size_t)t * BATCH + m) * NG + n] = h;
      }
    }
  } else {
    // ---------------- decoder for step t-1 ----------------
    int dt = t - 1;
    if (dt < 0) return;
    int wave  = ((blockIdx.x - STEP_BLOCKS) * blockDim.x + threadIdx.x) >> 5;
    int mtile = wave >> 3;         // NP/64 = 8 n-tiles per m-tile
    int nt    = wave & 7;
    int mBase = mtile * 16, nBase = nt * 64;
    v8f acc[4] = {};
    gemm_tile<NG>(hcur, Wdecf, mBase, nBase, lane, acc);
#pragma unroll
    for (int s = 0; s < 4; ++s) {
      int n = nBase + s * 16 + lane16;
#pragma unroll
      for (int r = 0; r < 8; ++r) {
        int m = mBase + half * 8 + r;
        if (m < BATCH)
          logits[((size_t)dt * BATCH + m) * NP + n] = acc[s][r];
      }
    }
  }
}

// ---- Standalone decoder (final step) -------------------------------------
__global__ __launch_bounds__(256) void decoder_kernel(
    const __bf16* __restrict__ h, const __bf16* __restrict__ Wdecf,
    float* __restrict__ logits, int t) {
  int wave = (blockIdx.x * blockDim.x + threadIdx.x) >> 5;
  int lane = threadIdx.x & 31;
  int mtile = wave >> 3;
  int nt    = wave & 7;
  int mBase = mtile * 16, nBase = nt * 64;
  v8f acc[4] = {};
  gemm_tile<NG>(h, Wdecf, mBase, nBase, lane, acc);
  int lane16 = lane & 15, half = lane >> 4;
#pragma unroll
  for (int s = 0; s < 4; ++s) {
    int n = nBase + s * 16 + lane16;
#pragma unroll
    for (int r = 0; r < 8; ++r) {
      int m = mBase + half * 8 + r;
      if (m < BATCH)
        logits[((size_t)t * BATCH + m) * NP + n] = acc[s][r];
    }
  }
}

// ---------------------------------------------------------------------------
extern "C" void kernel_launch(void* const* d_in, const int* in_sizes, int n_in,
                              void* d_out, int out_size, void* d_ws,
                              size_t ws_size, hipStream_t stream) {
  const float* velocity = (const float*)d_in[0];  // (SEQ, B, 2)
  const float* init_pc  = (const float*)d_in[1];  // (B, NP)
  const float* W_enc    = (const float*)d_in[2];  // (NG, NP)
  const float* W_ih     = (const float*)d_in[3];  // (NG, 2)
  const float* W_hh     = (const float*)d_in[4];  // (NG, NG)
  const float* W_dec    = (const float*)d_in[5];  // (NP, NG)

  float* logits = (float*)d_out;                              // SEQ*B*NP
  float* gout   = (float*)d_out + (size_t)SEQ * BATCH * NP;   // SEQ*B*NG

  // Workspace carve-out (256B aligned): ~45.6 MB total
  char* ws = (char*)d_ws;
  auto carve = [&](size_t bytes) {
    char* p = ws;
    ws += (bytes + 255) & ~(size_t)255;
    return p;
  };
  __bf16* Whhf  = (__bf16*)carve(sizeof(__bf16) * (size_t)NG * NG);
  __bf16* Wencf = (__bf16*)carve(sizeof(__bf16) * (size_t)NG * NP);
  __bf16* Wdecf = (__bf16*)carve(sizeof(__bf16) * (size_t)NP * NG);
  __bf16* pcb   = (__bf16*)carve(sizeof(__bf16) * (size_t)MPAD * NP);
  __bf16* hA    = (__bf16*)carve(sizeof(__bf16) * (size_t)MPAD * NG);
  __bf16* hB    = (__bf16*)carve(sizeof(__bf16) * (size_t)MPAD * NG);

  // 1) Pre-format weights into WMMA-B bf16 block layout (pair stores)
  {
    int n = (NG * NG) >> 1;
    fmt_weight_bf16<<<(n + 255) / 256, 256, 0, stream>>>(W_hh, Whhf, NG, NG);
  }
  {
    int n = (NG * NP) >> 1;
    fmt_weight_bf16<<<(n + 255) / 256, 256, 0, stream>>>(W_enc, Wencf, NG, NP);
  }
  {
    int n = (NP * NG) >> 1;
    fmt_weight_bf16<<<(n + 255) / 256, 256, 0, stream>>>(W_dec, Wdecf, NP, NG);
  }
  {
    int n = MPAD * NP;
    pad_pc_bf16<<<(n + 255) / 256, 256, 0, stream>>>(init_pc, pcb);
  }

  // 2) Encoder: h0 = init_pc @ W_enc^T   (13*64 = 832 waves)
  encoder_kernel<<<STEP_BLOCKS, 256, 0, stream>>>(pcb, Wencf, hA);

  // 3) Recurrence with fused previous-step decoder, ping-pong buffers
  __bf16* cur = hA;
  __bf16* nxt = hB;
  for (int t = 0; t < SEQ; ++t) {
    rnn_step_fused_kernel<<<STEP_BLOCKS + DEC_BLOCKS, 256, 0, stream>>>(
        cur, Whhf, Wdecf, velocity, W_ih, nxt, gout, logits, t);
    __bf16* tmp = cur; cur = nxt; nxt = tmp;
  }
  // Final decoder: logits[SEQ-1] from the last hidden state
  decoder_kernel<<<DEC_BLOCKS, 256, 0, stream>>>(cur, Wdecf, logits, SEQ - 1);
}
